// HolE_59931973648705
// MI455X (gfx1250) — matrix operand order repaired and checked by reading, not compiled, not run on hardware
//
#include <hip/hip_runtime.h>

typedef _Float16 v16h __attribute__((ext_vector_type(16)));
typedef float    v8f  __attribute__((ext_vector_type(8)));

#define HID      256
#define BATCH    8192
#define NEGS     16
#define NSAMP    (BATCH + BATCH * NEGS)      // 139264
#define NTILES   (NSAMP / 16)                // 8704
#define POSTILES (BATCH / 16)                // 512
#define TWO_PI_OVER_N 0.0245436926061702596f // 2*pi/256

// ---------------- workspace layout (bytes) ----------------
#define WS_BC   0                            // 65536 f16  (swizzled cos table)
#define WS_BS   131072                       // 65536 f16  (swizzled -sin table)
#define WS_RC   262144                       // 1000*256 f32
#define WS_RS   1286144                      // 1000*256 f32
#define WS_PSC  2310144                      // 8192 f32 positive scores
#define WS_NSC  2342912                      // 131072 f32 negative scores
#define WS_PART 2867200                      // 32 f32 partial sums

// ---------------------------------------------------------------------------
// Build swizzled DFT B-matrices (f16), laid out exactly as WMMA B operands:
// index = ((fc*8 + kc)*512) + lane*16 + j ;  lane<16: n=lane,k_local=(j<8 ? j : 8+j)
//                                            lane>=16: n=lane-16,k_local=(j<8 ? 8+j : 16+j)
// B[i][f] = cos(2*pi*((i*f)&255)/256),  Bs = -sin(...)
// ---------------------------------------------------------------------------
__global__ __launch_bounds__(256) void k_btab(_Float16* __restrict__ wsBc,
                                              _Float16* __restrict__ wsBs) {
    int gid  = blockIdx.x * 256 + threadIdx.x;      // 0..65535
    int j    = gid & 15;
    int lane = (gid >> 4) & 31;
    int kc   = (gid >> 9) & 7;
    int fc   = gid >> 12;
    int i    = kc * 32 + ((lane < 16) ? 0 : 8) + (j & 7) + ((j & 8) << 1);
    int f    = fc * 16 + (lane & 15);
    int k    = (i * f) & 255;
    float ang = (float)k * TWO_PI_OVER_N;
    wsBc[gid] = (_Float16)__cosf(ang);
    wsBs[gid] = (_Float16)(-__sinf(ang));
}

// ---------------------------------------------------------------------------
// Per-relation: L2 normalize (eps like reference) then f32 DFT.
// Rc[f] = s*sum_i r[i]*cos(2pi f i/256),  Rs[f] = s*sum_i r[i]*(-sin(...))
// ---------------------------------------------------------------------------
__global__ __launch_bounds__(256) void k_rel(const float* __restrict__ rel,
                                             float* __restrict__ wsRc,
                                             float* __restrict__ wsRs) {
    __shared__ float ctab[256], stab[256], row[256], wsum[8];
    const int tid = threadIdx.x;
    const int b   = blockIdx.x;
    float ang = (float)tid * TWO_PI_OVER_N;
    ctab[tid] = __cosf(ang);
    stab[tid] = -__sinf(ang);
    float rv  = rel[b * HID + tid];
    row[tid]  = rv;
    float ss = rv * rv;
    ss += __shfl_xor(ss, 1, 32);  ss += __shfl_xor(ss, 2, 32);
    ss += __shfl_xor(ss, 4, 32);  ss += __shfl_xor(ss, 8, 32);
    ss += __shfl_xor(ss, 16, 32);
    if ((tid & 31) == 0) wsum[tid >> 5] = ss;
    __syncthreads();
    float tot = 0.f;
#pragma unroll
    for (int w = 0; w < 8; ++w) tot += wsum[w];
    float scale = rsqrtf(fmaxf(tot, 1e-12f));
    float ac = 0.f, as = 0.f;
    int k = 0;
    for (int i = 0; i < 256; ++i) {
        float v = row[i];
        ac = fmaf(v, ctab[k], ac);
        as = fmaf(v, stab[k], as);
        k = (k + tid) & 255;               // k = (i*tid) mod 256
    }
    wsRc[b * HID + tid] = ac * scale;
    wsRs[b * HID + tid] = as * scale;
}

// ---------------------------------------------------------------------------
// Main kernel: WMMA spectral scoring.
//   score_pre = (1/256) * sum_f (Hc*Tc + Hs*Ts)*Rc + (Hc*Ts - Hs*Tc)*Rs
//   score     = -sigmoid(score_pre)
// One wave handles 16 samples; f in 16 chunks of 16; K=256 via 8 WMMA steps.
// ---------------------------------------------------------------------------
__global__ __launch_bounds__(128) void k_main(
        const int* __restrict__ pos_h, const int* __restrict__ pos_t,
        const int* __restrict__ pos_r, const int* __restrict__ neg_h,
        const int* __restrict__ neg_t, const int* __restrict__ neg_r,
        const float* __restrict__ ent,
        const _Float16* __restrict__ wsBc, const _Float16* __restrict__ wsBs,
        const float* __restrict__ wsRc, const float* __restrict__ wsRs,
        float* __restrict__ posS, float* __restrict__ negS, int numSlots) {
    __shared__ __align__(32) _Float16 lBc[4096];   // 8KB: one f-chunk of Bc
    __shared__ __align__(32) _Float16 lBs[4096];   // 8KB: one f-chunk of Bs

    const int tid    = threadIdx.x;
    const int wave   = tid >> 5;
    const int lane   = tid & 31;
    const int lane16 = lane & 15;
    const bool hiH   = lane >= 16;
    const int slot   = blockIdx.x * 4 + wave;
    const int iters  = (NTILES + numSlots - 1) / numSlots;

    for (int it = 0; it < iters; ++it) {
        const int tile = slot + it * numSlots;
        const bool valid = tile < NTILES;

        int rIdx = 0;
        v16h ah[8], at[8];
        if (valid) {
            int hIdx, tIdx;
            if (tile < POSTILES) {
                int s = tile * 16 + lane16;
                hIdx = pos_h[s]; tIdx = pos_t[s]; rIdx = pos_r[s];
            } else {
                int s = (tile - POSTILES) * 16 + lane16;
                hIdx = neg_h[s]; tIdx = neg_t[s]; rIdx = neg_r[s];
            }
            const float* hrow = ent + (size_t)hIdx * HID;
            const float* trow = ent + (size_t)tIdx * HID;
            const int kb = hiH ? 8 : 0;
#pragma unroll
            for (int c = 0; c < 8; ++c) {
                int k0 = c * 32 + kb;
                float4 a0 = *(const float4*)(hrow + k0);
                float4 a1 = *(const float4*)(hrow + k0 + 4);
                float4 a2 = *(const float4*)(hrow + k0 + 16);
                float4 a3 = *(const float4*)(hrow + k0 + 20);
                ah[c][0]=(_Float16)a0.x; ah[c][1]=(_Float16)a0.y; ah[c][2]=(_Float16)a0.z; ah[c][3]=(_Float16)a0.w;
                ah[c][4]=(_Float16)a1.x; ah[c][5]=(_Float16)a1.y; ah[c][6]=(_Float16)a1.z; ah[c][7]=(_Float16)a1.w;
                ah[c][8]=(_Float16)a2.x; ah[c][9]=(_Float16)a2.y; ah[c][10]=(_Float16)a2.z; ah[c][11]=(_Float16)a2.w;
                ah[c][12]=(_Float16)a3.x; ah[c][13]=(_Float16)a3.y; ah[c][14]=(_Float16)a3.z; ah[c][15]=(_Float16)a3.w;
                float4 b0 = *(const float4*)(trow + k0);
                float4 b1 = *(const float4*)(trow + k0 + 4);
                float4 b2 = *(const float4*)(trow + k0 + 16);
                float4 b3 = *(const float4*)(trow + k0 + 20);
                at[c][0]=(_Float16)b0.x; at[c][1]=(_Float16)b0.y; at[c][2]=(_Float16)b0.z; at[c][3]=(_Float16)b0.w;
                at[c][4]=(_Float16)b1.x; at[c][5]=(_Float16)b1.y; at[c][6]=(_Float16)b1.z; at[c][7]=(_Float16)b1.w;
                at[c][8]=(_Float16)b2.x; at[c][9]=(_Float16)b2.y; at[c][10]=(_Float16)b2.z; at[c][11]=(_Float16)b2.w;
                at[c][12]=(_Float16)b3.x; at[c][13]=(_Float16)b3.y; at[c][14]=(_Float16)b3.z; at[c][15]=(_Float16)b3.w;
            }
        }

        float sacc[8];
#pragma unroll
        for (int r = 0; r < 8; ++r) sacc[r] = 0.f;

#pragma unroll 1
        for (int fc = 0; fc < 16; ++fc) {
            __syncthreads();
            {   // cooperative stage of this f-chunk's B tiles (8KB + 8KB)
                const uint4* srcC = (const uint4*)(wsBc + fc * 4096);
                const uint4* srcS = (const uint4*)(wsBs + fc * 4096);
                uint4* dstC = (uint4*)&lBc[0];
                uint4* dstS = (uint4*)&lBs[0];
#pragma unroll
                for (int r = 0; r < 4; ++r) {
                    dstC[tid + r * 128] = srcC[tid + r * 128];
                    dstS[tid + r * 128] = srcS[tid + r * 128];
                }
            }
            __syncthreads();
            if (valid) {
                v8f hc = {}, hs = {}, tc = {}, ts = {};
#pragma unroll
                for (int kc = 0; kc < 8; ++kc) {
                    v16h bc = *(const v16h*)&lBc[kc * 512 + lane * 16];
                    v16h bs = *(const v16h*)&lBs[kc * 512 + lane * 16];
                    hc = __builtin_amdgcn_wmma_f32_16x16x32_f16(false, ah[kc], false, bc, (short)0, hc, false, false);
                    hs = __builtin_amdgcn_wmma_f32_16x16x32_f16(false, ah[kc], false, bs, (short)0, hs, false, false);
                    tc = __builtin_amdgcn_wmma_f32_16x16x32_f16(false, at[kc], false, bc, (short)0, tc, false, false);
                    ts = __builtin_amdgcn_wmma_f32_16x16x32_f16(false, at[kc], false, bs, (short)0, ts, false, false);
                }
                const int f = fc * 16 + lane16;
#pragma unroll
                for (int r = 0; r < 8; ++r) {
                    int srcLane = r + (hiH ? 8 : 0);
                    int rl = __shfl(rIdx, srcLane, 32);
                    float rc = wsRc[rl * HID + f];
                    float rs = wsRs[rl * HID + f];
                    float vhc = hc[r], vhs = hs[r], vtc = tc[r], vts = ts[r];
                    sacc[r] += (vhc * vtc + vhs * vts) * rc + (vhc * vts - vhs * vtc) * rs;
                }
            }
        }

        if (valid) {
#pragma unroll
            for (int r = 0; r < 8; ++r) {
                float x = sacc[r];
                x += __shfl_xor(x, 1, 32);
                x += __shfl_xor(x, 2, 32);
                x += __shfl_xor(x, 4, 32);
                x += __shfl_xor(x, 8, 32);
                if (lane16 == 0) {
                    float pre = x * (1.0f / 256.0f);
                    float sc  = -1.0f / (1.0f + __expf(-pre));
                    int sIT   = r + (hiH ? 8 : 0);
                    if (tile < POSTILES) posS[tile * 16 + sIT] = sc;
                    else                 negS[(tile - POSTILES) * 16 + sIT] = sc;
                }
            }
        }
    }
}

// ---------------------------------------------------------------------------
// loss stage 1: per-b term = relu(p - mean16(n) + 1); block-reduce -> partials
// ---------------------------------------------------------------------------
__global__ __launch_bounds__(256) void k_loss1(const float* __restrict__ posS,
                                               const float* __restrict__ negS,
                                               float* __restrict__ part) {
    __shared__ float wred[8];
    int b = blockIdx.x * 256 + threadIdx.x;   // 0..8191
    float nm = 0.f;
#pragma unroll
    for (int j = 0; j < 16; ++j) nm += negS[b * 16 + j];
    nm *= (1.0f / 16.0f);
    float s = fmaxf(posS[b] - nm + 1.0f, 0.0f);
    s += __shfl_xor(s, 1, 32);  s += __shfl_xor(s, 2, 32);
    s += __shfl_xor(s, 4, 32);  s += __shfl_xor(s, 8, 32);
    s += __shfl_xor(s, 16, 32);
    if ((threadIdx.x & 31) == 0) wred[threadIdx.x >> 5] = s;
    __syncthreads();
    if (threadIdx.x == 0) {
        float t = 0.f;
#pragma unroll
        for (int w = 0; w < 8; ++w) t += wred[w];
        part[blockIdx.x] = t;
    }
}

__global__ void k_loss2(const float* __restrict__ part, float* __restrict__ out) {
    float v = part[threadIdx.x];              // 32 partials
    v += __shfl_xor(v, 1, 32);  v += __shfl_xor(v, 2, 32);
    v += __shfl_xor(v, 4, 32);  v += __shfl_xor(v, 8, 32);
    v += __shfl_xor(v, 16, 32);
    if (threadIdx.x == 0) out[0] = v;
}

// ---------------------------------------------------------------------------
extern "C" void kernel_launch(void* const* d_in, const int* in_sizes, int n_in,
                              void* d_out, int out_size, void* d_ws, size_t ws_size,
                              hipStream_t stream) {
    const int*   pos_h = (const int*)d_in[0];
    const int*   pos_t = (const int*)d_in[1];
    const int*   pos_r = (const int*)d_in[2];
    const int*   neg_h = (const int*)d_in[3];
    const int*   neg_t = (const int*)d_in[4];
    const int*   neg_r = (const int*)d_in[5];
    const float* ent   = (const float*)d_in[6];
    const float* rel   = (const float*)d_in[7];

    char* ws = (char*)d_ws;
    _Float16* wsBc = (_Float16*)(ws + WS_BC);
    _Float16* wsBs = (_Float16*)(ws + WS_BS);
    float* wsRc = (float*)(ws + WS_RC);
    float* wsRs = (float*)(ws + WS_RS);
    float* posS = (float*)(ws + WS_PSC);
    float* negS = (float*)(ws + WS_NSC);
    float* part = (float*)(ws + WS_PART);

    k_btab<<<256, 256, 0, stream>>>(wsBc, wsBs);
    k_rel <<<1000, 256, 0, stream>>>(rel, wsRc, wsRs);
    const int blocks = 544;                    // 2176 wave-slots -> 4 tiles each
    k_main<<<blocks, 128, 0, stream>>>(pos_h, pos_t, pos_r, neg_h, neg_t, neg_r,
                                       ent, wsBc, wsBs, wsRc, wsRs,
                                       posS, negS, blocks * 4);
    k_loss1<<<32, 256, 0, stream>>>(posS, negS, part);
    k_loss2<<<1, 32, 0, stream>>>(part, (float*)d_out);
}
